// LSTM_Attention_LSTM_2353642078767
// MI455X (gfx1250) — matrix operand order repaired and checked by reading, not compiled, not run on hardware
//
#include <hip/hip_runtime.h>
#include <hip/hip_bf16.h>
#include <math.h>

// Problem dims (fixed by the reference)
#define B_   64
#define T_   256
#define D_   64
#define H_   512
#define G4H  2048   // 4*H

typedef __attribute__((ext_vector_type(16))) _Float16 v16h;
typedef __attribute__((ext_vector_type(8)))  _Float16 v8h;
typedef __attribute__((ext_vector_type(8)))  float    v8f;

union AFrag { v16h v; v8h h[2]; };

__device__ __forceinline__ v8f wmma_f16f32(v16h a, v16h b, v8f c) {
  // (neg_a, A, neg_b, B, c_mod, C, reuse_a, reuse_b)
  return __builtin_amdgcn_wmma_f32_16x16x32_f16(false, a, false, b, (short)0, c,
                                                false, false);
}

// ---------------------------------------------------------------------------
// Fragment loaders (wave32; A/B share a contiguous k order, C/D as in ISA 7.12.2)
// A 16x32 f16: lane L -> row = L&15; kgrp = L>>4; halves [0..15] = K = kgrp*16+j
//              (two 16B contiguous loads)
// B 32x16 f16: lane L -> col = L&15; halves [0..15] = K = (L>>4)*16 + j
//              B[k][n] == Wrow[n][k] so this is one contiguous 32B load
// C/D f32 16x16: lane L -> col = L&15; vgpr r -> row = r + 8*(L>>4)
// ---------------------------------------------------------------------------

__device__ __forceinline__ v16h load_afrag(const _Float16* rowptr, int k0, int kg) {
  AFrag a;
  a.h[0] = *(const v8h*)(rowptr + k0 + kg * 16);
  a.h[1] = *(const v8h*)(rowptr + k0 + kg * 16 + 8);
  return a.v;
}

__device__ __forceinline__ v16h load_bfrag(const _Float16* wrow, int k0, int kg) {
  return *(const v16h*)(wrow + k0 + kg * 16);
}

// ---------------------------------------------------------------------------
// gates[64,2048] = A1[64,K1]@W1^T + A2[64,K2]@W2^T + bih + bhh
// 2x2 register-blocked: each wave owns a 32x32 output block (4 WMMA tiles),
// 128 waves total (2 M-supertiles x 64 N-supertiles).
// ---------------------------------------------------------------------------
__global__ void k_gemm_gates(const _Float16* __restrict__ A1, int lda1, int K1,
                             const _Float16* __restrict__ W1,
                             const _Float16* __restrict__ A2, int K2,
                             const _Float16* __restrict__ W2,
                             const float* __restrict__ bih,
                             const float* __restrict__ bhh,
                             float* __restrict__ gates) {
  const int lane = threadIdx.x & 31;
  const int wave = blockIdx.x * (blockDim.x >> 5) + (threadIdx.x >> 5);
  const int mst = wave >> 6;        // 2 M-supertiles (M=64)
  const int nst = wave & 63;        // 64 N-supertiles (N=2048)
  const int m0 = mst << 5, n0 = nst << 5;
  const int rA = lane & 15;
  const int kg = lane >> 4;
  const int colB0 = n0 + (lane & 15);
  const int colB1 = colB0 + 16;

  v8f a00 = {}, a01 = {}, a10 = {}, a11 = {};

  {
    const _Float16* ar0 = A1 + (size_t)(m0 + rA) * lda1;
    const _Float16* ar1 = A1 + (size_t)(m0 + 16 + rA) * lda1;
    const _Float16* w0 = W1 + (size_t)colB0 * K1;
    const _Float16* w1 = W1 + (size_t)colB1 * K1;
    for (int k0 = 0; k0 < K1; k0 += 32) {
      v16h f0 = load_afrag(ar0, k0, kg);
      v16h f1 = load_afrag(ar1, k0, kg);
      v16h g0 = load_bfrag(w0, k0, kg);
      v16h g1 = load_bfrag(w1, k0, kg);
      a00 = wmma_f16f32(f0, g0, a00);
      a01 = wmma_f16f32(f0, g1, a01);
      a10 = wmma_f16f32(f1, g0, a10);
      a11 = wmma_f16f32(f1, g1, a11);
    }
  }
  {
    const _Float16* ar0 = A2 + (size_t)(m0 + rA) * K2;
    const _Float16* ar1 = A2 + (size_t)(m0 + 16 + rA) * K2;
    const _Float16* w0 = W2 + (size_t)colB0 * K2;
    const _Float16* w1 = W2 + (size_t)colB1 * K2;
    for (int k0 = 0; k0 < K2; k0 += 32) {
      if (k0 + 256 < K2) {
        __builtin_prefetch(w0 + k0 + 256, 0, 1);
        __builtin_prefetch(w1 + k0 + 256, 0, 1);
      }
      v16h f0 = load_afrag(ar0, k0, kg);
      v16h f1 = load_afrag(ar1, k0, kg);
      v16h g0 = load_bfrag(w0, k0, kg);
      v16h g1 = load_bfrag(w1, k0, kg);
      a00 = wmma_f16f32(f0, g0, a00);
      a01 = wmma_f16f32(f0, g1, a01);
      a10 = wmma_f16f32(f1, g0, a10);
      a11 = wmma_f16f32(f1, g1, a11);
    }
  }
  const float bs0 = bih[colB0] + bhh[colB0];
  const float bs1 = bih[colB1] + bhh[colB1];
#pragma unroll
  for (int r = 0; r < 8; ++r) {
    const size_t row0 = (size_t)(m0 + r + 8 * kg) * G4H;
    const size_t row1 = (size_t)(m0 + 16 + r + 8 * kg) * G4H;
    gates[row0 + colB0] = a00[r] + bs0;
    gates[row0 + colB1] = a01[r] + bs1;
    gates[row1 + colB0] = a10[r] + bs0;
    gates[row1 + colB1] = a11[r] + bs1;
  }
}

// ---------------------------------------------------------------------------
// Generic out[M,N] = A[M,K]@W[N,K]^T (+bias[N]); 2x2 register-blocked.
// Grid must supply exactly (M/32)*(N/32) waves; nst_cnt = N/32.
// ---------------------------------------------------------------------------
template <typename OutT>
__global__ void k_gemm(const _Float16* __restrict__ A, int lda,
                       const _Float16* __restrict__ W, int K, int nst_cnt,
                       const float* __restrict__ bias,
                       OutT* __restrict__ out, int ldo) {
  const int lane = threadIdx.x & 31;
  const int wave = blockIdx.x * (blockDim.x >> 5) + (threadIdx.x >> 5);
  const int mst = wave / nst_cnt;
  const int nst = wave % nst_cnt;
  const int m0 = mst << 5, n0 = nst << 5;
  const int rA = lane & 15;
  const int kg = lane >> 4;
  const int colB0 = n0 + (lane & 15);
  const int colB1 = colB0 + 16;

  v8f a00 = {}, a01 = {}, a10 = {}, a11 = {};
  const _Float16* ar0 = A + (size_t)(m0 + rA) * lda;
  const _Float16* ar1 = A + (size_t)(m0 + 16 + rA) * lda;
  const _Float16* w0 = W + (size_t)colB0 * K;
  const _Float16* w1 = W + (size_t)colB1 * K;
  for (int k0 = 0; k0 < K; k0 += 32) {
    if (k0 + 256 < K) {
      __builtin_prefetch(w0 + k0 + 256, 0, 1);
      __builtin_prefetch(w1 + k0 + 256, 0, 1);
    }
    v16h f0 = load_afrag(ar0, k0, kg);
    v16h f1 = load_afrag(ar1, k0, kg);
    v16h g0 = load_bfrag(w0, k0, kg);
    v16h g1 = load_bfrag(w1, k0, kg);
    a00 = wmma_f16f32(f0, g0, a00);
    a01 = wmma_f16f32(f0, g1, a01);
    a10 = wmma_f16f32(f1, g0, a10);
    a11 = wmma_f16f32(f1, g1, a11);
  }
  const float bs0 = bias ? bias[colB0] : 0.f;
  const float bs1 = bias ? bias[colB1] : 0.f;
#pragma unroll
  for (int r = 0; r < 8; ++r) {
    const size_t row0 = (size_t)(m0 + r + 8 * kg) * ldo;
    const size_t row1 = (size_t)(m0 + 16 + r + 8 * kg) * ldo;
    out[row0 + colB0] = (OutT)(a00[r] + bs0);
    out[row0 + colB1] = (OutT)(a01[r] + bs1);
    out[row1 + colB0] = (OutT)(a10[r] + bs0);
    out[row1 + colB1] = (OutT)(a11[r] + bs1);
  }
}

// ---------------------------------------------------------------------------
// Elementwise LSTM cell. gates [64,2048] (i|f|g|o blocks of 512).
// Updates c(f32), h(f16); optionally mirrors h into enc_hs[:,t,:].
// ---------------------------------------------------------------------------
__device__ __forceinline__ float sigm(float x) { return 1.f / (1.f + __expf(-x)); }

__global__ void k_lstm_cell(const float* __restrict__ gates,
                            float* __restrict__ c,
                            _Float16* __restrict__ h,
                            _Float16* __restrict__ hs_t, int hs_stride) {
  const int idx = blockIdx.x * blockDim.x + threadIdx.x;  // 0..32767
  const int b = idx >> 9, n = idx & 511;
  const float* g = gates + (size_t)b * G4H;
  const float gi = g[n], gf = g[512 + n], gg = g[1024 + n], go = g[1536 + n];
  const float cc = sigm(gf) * c[idx] + sigm(gi) * tanhf(gg);
  const float hh = sigm(go) * tanhf(cc);
  c[idx] = cc;
  h[idx] = (_Float16)hh;
  if (hs_t) hs_t[(size_t)b * hs_stride + n] = (_Float16)hh;
}

// ---------------------------------------------------------------------------
// Fused per-decoder-step attention. One block per batch row (64 x 256thr).
// score_t = v . tanh(enc_proj[b,t,:] + hproj[b,:]); softmax over t;
// context = sum_t w_t * enc_hs[b,t,:]; out = fc.ctx+fcb; dec_in = out*o2h+o2hb
// ---------------------------------------------------------------------------
__global__ void k_attn_step(const _Float16* __restrict__ enc_proj,
                            const _Float16* __restrict__ enc_hs,
                            const float* __restrict__ hproj,
                            const float* __restrict__ vW, const float* __restrict__ vb,
                            const float* __restrict__ fcW, const float* __restrict__ fcb,
                            const float* __restrict__ o2hW, const float* __restrict__ o2hb,
                            float* __restrict__ out, int t,
                            _Float16* __restrict__ dec_in) {
  __shared__ float hp[H_];
  __shared__ float vv[H_];
  __shared__ float sc[T_];
  __shared__ float red[256];
  const int b = blockIdx.x;
  const int j = threadIdx.x;  // 0..255

  hp[j] = hproj[b * H_ + j];
  hp[j + 256] = hproj[b * H_ + j + 256];
  vv[j] = vW[j];
  vv[j + 256] = vW[j + 256];
  __syncthreads();

  // score for timestep j
  const _Float16* ep = enc_proj + ((size_t)b * T_ + j) * H_;
  float acc = 0.f;
  for (int h = 0; h < H_; ++h) {
    const float e = tanhf((float)ep[h] + hp[h]);
    acc += vv[h] * e;
  }
  sc[j] = acc + vb[0];
  __syncthreads();

  // softmax over 256 timesteps
  red[j] = sc[j];
  __syncthreads();
  for (int s = 128; s > 0; s >>= 1) {
    if (j < s) red[j] = fmaxf(red[j], red[j + s]);
    __syncthreads();
  }
  const float mx = red[0];
  __syncthreads();
  const float ex = __expf(sc[j] - mx);
  red[j] = ex;
  __syncthreads();
  for (int s = 128; s > 0; s >>= 1) {
    if (j < s) red[j] += red[j + s];
    __syncthreads();
  }
  const float inv = 1.f / red[0];
  __syncthreads();
  sc[j] = ex * inv;
  __syncthreads();

  // context for columns j and j+256 (coalesced across threads per timestep)
  float c0 = 0.f, c1 = 0.f;
  const _Float16* eh = enc_hs + (size_t)b * T_ * H_;
  for (int tt = 0; tt < T_; ++tt) {
    const float w = sc[tt];
    c0 += w * (float)eh[(size_t)tt * H_ + j];
    c1 += w * (float)eh[(size_t)tt * H_ + j + 256];
  }

  // out = fc . context + fcb   (block reduce)
  red[j] = fcW[j] * c0 + fcW[j + 256] * c1;
  __syncthreads();
  for (int s = 128; s > 0; s >>= 1) {
    if (j < s) red[j] += red[j + s];
    __syncthreads();
  }
  const float ov = red[0] + fcb[0];
  if (j == 0) out[(size_t)b * T_ + t] = ov;

  // feedback projection dec_in = ov * o2h_W + o2h_b
  dec_in[b * H_ + j] = (_Float16)(ov * o2hW[j] + o2hb[j]);
  dec_in[b * H_ + j + 256] = (_Float16)(ov * o2hW[j + 256] + o2hb[j + 256]);
}

// ---------------------------------------------------------------------------
// f32 -> f16 conversion helpers
// ---------------------------------------------------------------------------
__global__ void k_cvt(const float* __restrict__ s, _Float16* __restrict__ d, int n) {
  const int i = blockIdx.x * blockDim.x + threadIdx.x;
  if (i < n) d[i] = (_Float16)s[i];
}

// attn_W [512,1024] -> W1 = cols [0,512), W2 = cols [512,1024)
__global__ void k_split_attn(const float* __restrict__ W,
                             _Float16* __restrict__ W1,
                             _Float16* __restrict__ W2) {
  const int i = blockIdx.x * blockDim.x + threadIdx.x;  // 0..262143
  const int n = i >> 9, k = i & 511;
  W1[i] = (_Float16)W[n * 1024 + k];
  W2[i] = (_Float16)W[n * 1024 + 512 + k];
}

// ---------------------------------------------------------------------------
extern "C" void kernel_launch(void* const* d_in, const int* in_sizes, int n_in,
                              void* d_out, int out_size, void* d_ws, size_t ws_size,
                              hipStream_t stream) {
  (void)in_sizes; (void)n_in; (void)out_size; (void)ws_size;
  const float* x    = (const float*)d_in[0];
  const float* eWih = (const float*)d_in[1];
  const float* eWhh = (const float*)d_in[2];
  const float* ebih = (const float*)d_in[3];
  const float* ebhh = (const float*)d_in[4];
  const float* dWih = (const float*)d_in[5];
  const float* dWhh = (const float*)d_in[6];
  const float* dbih = (const float*)d_in[7];
  const float* dbhh = (const float*)d_in[8];
  const float* attnW = (const float*)d_in[9];
  const float* attnb = (const float*)d_in[10];
  const float* vW  = (const float*)d_in[11];
  const float* vb  = (const float*)d_in[12];
  const float* fcW = (const float*)d_in[13];
  const float* fcb = (const float*)d_in[14];
  const float* o2hW = (const float*)d_in[15];
  const float* o2hb = (const float*)d_in[16];
  float* out = (float*)d_out;

  // workspace carve-up (256B aligned)
  char* p = (char*)d_ws;
  auto alloc = [&](size_t bytes) -> char* {
    char* r = p;
    p += (bytes + 255) & ~(size_t)255;
    return r;
  };
  _Float16* xh   = (_Float16*)alloc((size_t)B_ * T_ * D_ * 2);
  _Float16* heWi = (_Float16*)alloc((size_t)G4H * D_ * 2);
  _Float16* heWh = (_Float16*)alloc((size_t)G4H * H_ * 2);
  _Float16* hdWi = (_Float16*)alloc((size_t)G4H * H_ * 2);
  _Float16* hdWh = (_Float16*)alloc((size_t)G4H * H_ * 2);
  _Float16* aW1  = (_Float16*)alloc((size_t)H_ * H_ * 2);
  _Float16* aW2  = (_Float16*)alloc((size_t)H_ * H_ * 2);
  _Float16* enc_hs   = (_Float16*)alloc((size_t)B_ * T_ * H_ * 2);
  _Float16* enc_proj = (_Float16*)alloc((size_t)B_ * T_ * H_ * 2);
  _Float16* hst  = (_Float16*)alloc((size_t)B_ * H_ * 2);   // h (enc then dec)
  float*    cst  = (float*)   alloc((size_t)B_ * H_ * 4);   // c (enc then dec)
  _Float16* dec_in = (_Float16*)alloc((size_t)B_ * H_ * 2);
  float*    gates  = (float*)  alloc((size_t)B_ * G4H * 4);
  float*    hproj  = (float*)  alloc((size_t)B_ * H_ * 4);

  // zero initial state
  hipMemsetAsync(hst, 0, (size_t)B_ * H_ * 2, stream);
  hipMemsetAsync(cst, 0, (size_t)B_ * H_ * 4, stream);
  hipMemsetAsync(dec_in, 0, (size_t)B_ * H_ * 2, stream);

  // one-time f16 conversions
  k_cvt<<<4096, 256, 0, stream>>>(x, xh, B_ * T_ * D_);
  k_cvt<<<512, 256, 0, stream>>>(eWih, heWi, G4H * D_);
  k_cvt<<<4096, 256, 0, stream>>>(eWhh, heWh, G4H * H_);
  k_cvt<<<4096, 256, 0, stream>>>(dWih, hdWi, G4H * H_);
  k_cvt<<<4096, 256, 0, stream>>>(dWhh, hdWh, G4H * H_);
  k_split_attn<<<1024, 256, 0, stream>>>(attnW, aW1, aW2);

  // -------- encoder: 256 sequential LSTM steps --------
  // gate GEMM: 2 x 64 supertiles = 128 waves -> 16 blocks x 8 waves
  for (int t = 0; t < T_; ++t) {
    k_gemm_gates<<<16, 256, 0, stream>>>(xh + (size_t)t * D_, T_ * D_, D_, heWi,
                                         hst, H_, heWh, ebih, ebhh, gates);
    k_lstm_cell<<<128, 256, 0, stream>>>(gates, cst, hst,
                                         enc_hs + (size_t)t * H_, T_ * H_);
  }

  // -------- enc_proj = enc_hs @ attn_W2^T + attn_b  (constant over decoder) --
  // M = B*T = 16384, N = 512 -> (512 x 16) supertiles = 8192 waves
  k_gemm<_Float16><<<1024, 256, 0, stream>>>(enc_hs, H_, aW2, H_, H_ / 32,
                                             attnb, enc_proj, H_);

  // -------- decoder: 256 sequential steps --------
  for (int t = 0; t < T_; ++t) {
    k_gemm_gates<<<16, 256, 0, stream>>>(dec_in, H_, H_, hdWi,
                                         hst, H_, hdWh, dbih, dbhh, gates);
    k_lstm_cell<<<128, 256, 0, stream>>>(gates, cst, hst, (_Float16*)nullptr, 0);
    // hproj = h @ attn_W1^T   (M=64,N=512 -> 2x16 supertiles = 32 waves)
    k_gemm<float><<<4, 256, 0, stream>>>(hst, H_, aW1, H_, H_ / 32,
                                         (const float*)nullptr, hproj, H_);
    k_attn_step<<<64, 256, 0, stream>>>(enc_proj, enc_hs, hproj, vW, vb,
                                        fcW, fcb, o2hW, o2hb, out, t, dec_in);
  }
}
